// LocalTrans_32796370272313
// MI455X (gfx1250) — matrix operand order repaired and checked by reading, not compiled, not run on hardware
//
#include <hip/hip_runtime.h>
#include <hip/hip_bf16.h>
#include <cmath>

typedef __attribute__((ext_vector_type(16))) _Float16 v16h;
typedef __attribute__((ext_vector_type(8)))  _Float16 v8h;
typedef __attribute__((ext_vector_type(8)))  float    v8f;

#define B_   2
#define N_   4096
#define D_   512
#define H_   8
#define DH_  64
#define WIN_ 512
#define L_   6
#define DFF_ 2048
#define M_   (B_ * N_)   // 8192

#if defined(__has_builtin)
#if __has_builtin(__builtin_amdgcn_global_load_async_to_lds_b128) && \
    __has_builtin(__builtin_amdgcn_s_wait_asynccnt)
#define ASYNC_LDS 1
#endif
#if __has_builtin(__builtin_amdgcn_tensor_load_to_lds) && \
    __has_builtin(__builtin_amdgcn_s_wait_tensorcnt)
#define HAVE_TDM 1
#endif
#endif

#if defined(HAVE_TDM)
#if defined(__has_include)
#if __has_include(<hip/amd_detail/amd_gfx1250_TDM.h>)
#define TDM_6ARG 1
#endif
#endif
#endif

#ifdef ASYNC_LDS
typedef int v4i __attribute__((vector_size(16)));
typedef __attribute__((address_space(1))) v4i* gas1;   // global int4*
typedef __attribute__((address_space(3))) v4i* las3;   // LDS int4*
#endif

#ifdef HAVE_TDM
typedef unsigned tdm_u4 __attribute__((ext_vector_type(4)));
typedef int      tdm_i4 __attribute__((ext_vector_type(4)));
typedef int      tdm_i8 __attribute__((ext_vector_type(8)));

// 2D tile DMA: global[rows][strideElems] -> LDS[rows][rowElems], 16-bit elements.
// D# per CDNA5 ISA §8.3/8.4: count=1, type=2("image"), data_size=1(2B),
// tensor_dim0=tile_dim0=rowElems, tensor_dim1=tile_dim1=rows, dim0_stride=strideElems.
static __device__ __forceinline__ void tdm_load_2d_f16(const void* gsrc, void* lds_dst,
                                                       unsigned rowElems, unsigned rows,
                                                       unsigned strideElems) {
  unsigned long long ga = (unsigned long long)gsrc;
  unsigned lo = (unsigned)(unsigned long long)(__attribute__((address_space(3))) char*)lds_dst;
  tdm_u4 g0;
  g0.x = 1u;                                  // [1:0] count=1 (valid), rest 0
  g0.y = lo;                                  // [63:32]  lds_addr (bytes)
  g0.z = (unsigned)ga;                        // [95:64]  global_addr[31:0]
  g0.w = (unsigned)(ga >> 32) | (2u << 30);   // [120:96] addr[56:32] | [127:126] type=2
  tdm_i8 g1;
  g1[0] = (1 << 16);                          // workgroup_mask=0, data_size=1 (2 bytes)
  g1[1] = (int)(rowElems << 16);              // [63:48] tensor_dim0 lo16
  g1[2] = (int)(rows << 16);                  // [79:64] dim0 hi=0, [95:80] tensor_dim1 lo16
  g1[3] = (int)(rowElems << 16);              // [111:96] dim1 hi=0, [127:112] tile_dim0
  g1[4] = (int)rows;                          // [143:128] tile_dim1, [159:144] tile_dim2=0
  g1[5] = (int)strideElems;                   // [207:160] tensor_dim0_stride lo32
  g1[6] = 0;
  g1[7] = 0;
  tdm_i4 z4 = {0, 0, 0, 0};
#ifdef TDM_6ARG
  tdm_i8 z8 = {0, 0, 0, 0, 0, 0, 0, 0};
  __builtin_amdgcn_tensor_load_to_lds(g0, g1, z4, z4, z8, 0);
#else
  __builtin_amdgcn_tensor_load_to_lds(g0, g1, z4, z4, 0);
#endif
}
#endif

static __device__ __forceinline__ v16h cat8(v8h a, v8h b) {
  return __builtin_shufflevector(a, b, 0,1,2,3,4,5,6,7,8,9,10,11,12,13,14,15);
}

// wave-internal LDS ordering (LDS ops of one wave complete in-order per DScnt)
static __device__ __forceinline__ void wave_lds_sync() {
  __builtin_amdgcn_fence(__ATOMIC_ACQ_REL, "wavefront");
  __builtin_amdgcn_wave_barrier();
  __builtin_amdgcn_fence(__ATOMIC_ACQ_REL, "wavefront");
}

// ------------------------------------------------ convert + transpose weights
__global__ void k_conv_tr(const float* __restrict__ src, _Float16* __restrict__ dst,
                          int K, int Nn) {
  int idx = blockIdx.x * 256 + threadIdx.x;
  int total = K * Nn;
  int stride = gridDim.x * 256;
  for (; idx < total; idx += stride) {
    int n = idx / K, k = idx - n * K;          // dst-linear => coalesced writes
    dst[idx] = (_Float16)src[(size_t)k * Nn + n];
  }
}

// ------------------------------------------------ layernorm -> f16
__global__ void k_layernorm_f16(const float* __restrict__ x, const float* __restrict__ g,
                                const float* __restrict__ b, _Float16* __restrict__ out) {
  const int row = blockIdx.x;
  const int t = threadIdx.x;
  const float* xr = x + (size_t)row * D_;
  float v0 = xr[t], v1 = xr[t + 256];

  __shared__ float red[8];
  float s = v0 + v1;
  #pragma unroll
  for (int m = 16; m >= 1; m >>= 1) s += __shfl_xor(s, m, 32);
  if ((t & 31) == 0) red[t >> 5] = s;
  __syncthreads();
  float mean = 0.f;
  #pragma unroll
  for (int i = 0; i < 8; ++i) mean += red[i];
  mean *= (1.0f / (float)D_);
  __syncthreads();

  float d0 = v0 - mean, d1 = v1 - mean;
  float vs = d0 * d0 + d1 * d1;
  #pragma unroll
  for (int m = 16; m >= 1; m >>= 1) vs += __shfl_xor(vs, m, 32);
  if ((t & 31) == 0) red[t >> 5] = vs;
  __syncthreads();
  float var = 0.f;
  #pragma unroll
  for (int i = 0; i < 8; ++i) var += red[i];
  var *= (1.0f / (float)D_);
  float rstd = rsqrtf(var + 1e-5f);

  out[(size_t)row * D_ + t]       = (_Float16)(d0 * rstd * g[t] + b[t]);
  out[(size_t)row * D_ + t + 256] = (_Float16)(d1 * rstd * g[t + 256] + b[t + 256]);
}

// ------------------------------------------------ WMMA GEMM (double-buffered, TDM/async)
// C[M,Nn] = A16[M,K] @ Bt16[Nn,K]^T ; block tile 128x128, 8 waves (4m x 2n),
// wave tile 32x64 (8 wmma / K-step), K-step 32. A/B tiles DMA'd by waves 0/1.
// EPI: 0 = f32 out; 1 = f32 out + (bias) + resid; 2 = f16 gelu(acc + bias);
//      3 = qkv epilogue: rotary(q,k) + q*0.125, f16 out (Nn == 1536)
template<int EPI>
__global__ void k_gemm_wmma(const _Float16* __restrict__ A, const _Float16* __restrict__ Bt,
                            const float* __restrict__ bias, const float* resid,
                            void* outv, int K, int Nn) {
  __shared__ __align__(16) _Float16 As[2][128 * 32];   // [m][k]
  __shared__ __align__(16) _Float16 Bs[2][128 * 32];   // [n][k]

  const int t = threadIdx.x;
  const int lane = t & 31, wv = t >> 5;
  const int wm = wv & 3, wn = wv >> 2;
  const bool hi = lane >= 16;
  const int l15 = lane & 15;
  const int mB = blockIdx.y * 128;
  const int nB = blockIdx.x * 128;

  v8f acc[2][4] = {};

  const int srow = t >> 1;            // 0..127
  const int koff = (t & 1) * 16;      // 0 / 16 halves
  const _Float16* gA = A  + (size_t)(mB + srow) * K + koff;
  const _Float16* gB = Bt + (size_t)(nB + srow) * K + koff;

  auto stage = [&](int k0, int buf) {
#ifdef HAVE_TDM
    // parallel DMA: wave 0 moves the A tile, wave 1 the B tile
    if (wv == 0)      tdm_load_2d_f16(A  + (size_t)mB * K + k0, &As[buf][0], 32, 128, (unsigned)K);
    else if (wv == 1) tdm_load_2d_f16(Bt + (size_t)nB * K + k0, &Bs[buf][0], 32, 128, (unsigned)K);
#else
    const _Float16* ga = gA + k0;
    const _Float16* gb = gB + k0;
    _Float16* sa = &As[buf][srow * 32 + koff];
    _Float16* sb = &Bs[buf][srow * 32 + koff];
#ifdef ASYNC_LDS
    __builtin_amdgcn_global_load_async_to_lds_b128((gas1)ga, (las3)sa, 0, 0);
    __builtin_amdgcn_global_load_async_to_lds_b128((gas1)(ga + 8), (las3)(sa + 8), 0, 0);
    __builtin_amdgcn_global_load_async_to_lds_b128((gas1)gb, (las3)sb, 0, 0);
    __builtin_amdgcn_global_load_async_to_lds_b128((gas1)(gb + 8), (las3)(sb + 8), 0, 0);
#else
    ((uint4*)sa)[0] = ((const uint4*)ga)[0];
    ((uint4*)sa)[1] = ((const uint4*)ga)[1];
    ((uint4*)sb)[0] = ((const uint4*)gb)[0];
    ((uint4*)sb)[1] = ((const uint4*)gb)[1];
    __builtin_prefetch(ga + 32, 0, 3);
    __builtin_prefetch(gb + 32, 0, 3);
#endif
#endif
  };

  stage(0, 0);
  const int steps = K >> 5;
  for (int it = 0; it < steps; ++it) {
    const int cur = it & 1;
#ifdef HAVE_TDM
    if (wv < 2) __builtin_amdgcn_s_wait_tensorcnt(0);   // own tile landed
#elif defined(ASYNC_LDS)
    __builtin_amdgcn_s_wait_asynccnt(0);
#endif
    __syncthreads();                        // data ready; everyone past tile it-1
    if (it + 1 < steps) stage((it + 1) << 5, cur ^ 1);

    const _Float16* as = As[cur];
    const _Float16* bs = Bs[cur];
    v16h af[2], bf[4];
    #pragma unroll
    for (int f = 0; f < 2; ++f) {
      // A frag 16x32: lanes0-15 row=l,K{0-7,16-23}; lanes16-31 row=l-16,K{8-15,24-31}
      int r = wm * 32 + f * 16 + l15;
      int off = hi ? 8 : 0;
      v8h p0 = *(const v8h*)&as[r * 32 + off];
      v8h p1 = *(const v8h*)&as[r * 32 + 16 + off];
      af[f] = cat8(p0, p1);
    }
    #pragma unroll
    for (int f = 0; f < 4; ++f) {
      // B frag 32x16: lanes0-15 col=l,K{0-7,16-23}; lanes16-31 col=l-16,K{8-15,24-31}
      int c = wn * 64 + f * 16 + l15;
      int boff = hi ? 8 : 0;
      v8h q0 = *(const v8h*)&bs[c * 32 + boff];
      v8h q1 = *(const v8h*)&bs[c * 32 + boff + 16];
      bf[f] = cat8(q0, q1);
    }
    #pragma unroll
    for (int fm = 0; fm < 2; ++fm)
      #pragma unroll
      for (int fn = 0; fn < 4; ++fn)
        acc[fm][fn] = __builtin_amdgcn_wmma_f32_16x16x32_f16(
            false, af[fm], false, bf[fn], (short)0, acc[fm][fn], false, false);
  }

  // epilogue: D layout — VGPR i: lanes0-15 (M=i,N=l), lanes16-31 (M=8+i,N=l-16)
  if constexpr (EPI == 3) {
    // qkv epilogue: cols [0,512)=q (rotary*0.125), [512,1024)=k (rotary), [1024,1536)=v.
    _Float16* outq = (_Float16*)outv;
    const bool  isQK   = nB < 1024;
    const float qscale = (nB < 512) ? 0.125f : 1.0f;
    #pragma unroll
    for (int fm = 0; fm < 2; ++fm) {
      #pragma unroll
      for (int fn = 0; fn < 2; ++fn) {
        int pr = fn * 16 + l15;               // rotary pair index 0..31
        float invf = __expf(-(float)pr * 0.28782313662f);  // ln(1e4)/32
        #pragma unroll
        for (int i = 0; i < 8; ++i) {
          int row = mB + wm * 32 + fm * 16 + i + (hi ? 8 : 0);
          int col = nB + wn * 64 + fn * 16 + l15;
          size_t idx = (size_t)row * 1536 + col;
          float x1 = acc[fm][fn][i];
          float x2 = acc[fm][fn + 2][i];
          if (isQK) {
            float fr = (float)(row & (N_ - 1)) * invf;
            float c = __cosf(fr), s = __sinf(fr);
            outq[idx]      = (_Float16)((x1 * c - x2 * s) * qscale);
            outq[idx + 32] = (_Float16)((x2 * c + x1 * s) * qscale);
          } else {
            outq[idx]      = (_Float16)x1;
            outq[idx + 32] = (_Float16)x2;
          }
        }
      }
    }
  } else {
    #pragma unroll
    for (int fm = 0; fm < 2; ++fm) {
      #pragma unroll
      for (int fn = 0; fn < 4; ++fn) {
        #pragma unroll
        for (int i = 0; i < 8; ++i) {
          int row = mB + wm * 32 + fm * 16 + i + (hi ? 8 : 0);
          int col = nB + wn * 64 + fn * 16 + l15;
          size_t idx = (size_t)row * Nn + col;
          float v = acc[fm][fn][i];
          if constexpr (EPI == 0) {
            ((float*)outv)[idx] = v;
          } else if constexpr (EPI == 1) {
            if (bias) v += bias[col];
            ((float*)outv)[idx] = v + resid[idx];
          } else {
            v += bias[col];
            float gl = 0.5f * v * (1.0f + erff(v * 0.70710678118f));
            ((_Float16*)outv)[idx] = (_Float16)gl;
          }
        }
      }
    }
  }
}

// ------------------------------------------------ local attention (transposed flash)
// qkv16: [b*N+n][1536] f16, ALREADY rotary'd, q pre-scaled by d^-1/2.
// simT = K @ Q^T and O^T = V^T @ P^T: softmax row (query) lives in the lane index.
// K and V tiles DMA'd by TDM (waves 0/1 in parallel); V transposed LDS->LDS.
// grid: 512 blocks = (b:2)(h:8)(w:8)(qb:4); 256 threads = 8 waves, 16 queries/wave.
__global__ void k_local_attn(const _Float16* __restrict__ qkv, _Float16* __restrict__ outh) {
  const int bid = blockIdx.x;
  const int qb = bid & 3;
  const int w  = (bid >> 2) & 7;
  const int h  = (bid >> 5) & 7;
  const int b  = bid >> 8;
  const int t = threadIdx.x;
  const int lane = t & 31, wv = t >> 5;
  const bool hi = lane >= 16;
  const int l15 = lane & 15;

  __shared__ __align__(16) _Float16 ks[64 * 64];     // [key][d]
  __shared__ __align__(16) _Float16 vs[64 * 64];     // [d][key]
#ifdef HAVE_TDM
  __shared__ __align__(16) _Float16 vraw[64 * 64];   // [key][d] (TDM landing buffer)
#endif
  __shared__ __align__(16) _Float16 ps[8 * 16 * 32]; // per-wave P^T as [q][key32]

  const int tok0 = w * WIN_ + qb * 128;
  const size_t rowbase = (size_t)b * N_;

  // ---- Q^T B-fragments straight from global (col = query = l15, rows = dims) ----
  v16h bq[2];
  {
    int q = tok0 + wv * 16 + l15;
    const _Float16* qp = qkv + (rowbase + q) * 1536 + h * 64;
    int off = hi ? 8 : 0;
    #pragma unroll
    for (int fd = 0; fd < 2; ++fd) {
      v8h p0 = *(const v8h*)&qp[fd * 32 + off];
      v8h p1 = *(const v8h*)&qp[fd * 32 + off + 16];
      bq[fd] = cat8(p0, p1);
    }
  }

  const int qa = tok0 + wv * 16 + l15;
  const int qaMin = tok0 + wv * 16;
  const int qaMax = tok0 + wv * 16 + 15;
  float mrow = -1e30f, lrow = 0.f;
  v8f oacc[4] = {};

  const int kaBase = (w - 1) * WIN_;               // keys span [kaBase, kaBase+1024)
  const int kt0 = (w == 0) ? (WIN_ >> 6) : 0;      // window 0: skip all-pad look-back
  const int ntiles = (WIN_ + qb * 128 + 128) >> 6;

  for (int kt = kt0; kt < ntiles; ++kt) {
    __syncthreads();                               // prior tile fully consumed
    const size_t tokK = rowbase + kaBase + kt * 64;
#ifdef HAVE_TDM
    if (wv == 0)
      tdm_load_2d_f16(qkv + tokK * 1536 + 512 + h * 64, ks, 64, 64, 1536);
    else if (wv == 1)
      tdm_load_2d_f16(qkv + tokK * 1536 + 1024 + h * 64, vraw, 64, 64, 1536);
    if (wv < 2) __builtin_amdgcn_s_wait_tensorcnt(0);
    __syncthreads();                               // ks + vraw landed
    // transpose vraw[key][d] -> vs[d][key] (packed b128 reads, b16 scatter writes)
    {
      int kr = t >> 2, qt = t & 3;
      v8h r0 = *(const v8h*)&vraw[kr * 64 + qt * 16];
      v8h r1 = *(const v8h*)&vraw[kr * 64 + qt * 16 + 8];
      #pragma unroll
      for (int j = 0; j < 8; ++j) vs[(qt * 16 + j) * 64 + kr]     = r0[j];
      #pragma unroll
      for (int j = 0; j < 8; ++j) vs[(qt * 16 + j + 8) * 64 + kr] = r1[j];
    }
    __syncthreads();                               // vs ready
#else
    // stage K tile -> ks[key][d]
    {
      int kr = t >> 2, qt = t & 3;
      const _Float16* kp = qkv + (tokK + kr) * 1536 + 512 + h * 64 + qt * 16;
      _Float16* sd = &ks[kr * 64 + qt * 16];
#ifdef ASYNC_LDS
      __builtin_amdgcn_global_load_async_to_lds_b128((gas1)kp, (las3)sd, 0, 0);
      __builtin_amdgcn_global_load_async_to_lds_b128((gas1)(kp + 8), (las3)(sd + 8), 0, 0);
#else
      ((uint4*)sd)[0] = ((const uint4*)kp)[0];
      ((uint4*)sd)[1] = ((const uint4*)kp)[1];
#endif
    }
    // stage V tile -> vs[d][key] (f16 gather, packed LDS stores)
    {
      int d = t >> 2, qt = t & 3;
      const _Float16* vp = qkv + (tokK + qt * 16) * 1536 + 1024 + h * 64 + d;
      v8h r0, r1;
      #pragma unroll
      for (int j = 0; j < 8; ++j)  r0[j] = vp[(size_t)j * 1536];
      #pragma unroll
      for (int j = 0; j < 8; ++j)  r1[j] = vp[(size_t)(j + 8) * 1536];
      *(v8h*)&vs[d * 64 + qt * 16]     = r0;
      *(v8h*)&vs[d * 64 + qt * 16 + 8] = r1;
    }
#ifdef ASYNC_LDS
    __builtin_amdgcn_s_wait_asynccnt(0);
#endif
    __syncthreads();
#endif

    #pragma unroll
    for (int kk = 0; kk < 2; ++kk) {         // 32-key sub-chunks
      int kaTile = kaBase + kt * 64 + kk * 32;
      if (kaTile > qaMax) continue;          // wave-uniform: fully causal-masked

      // ---- simT[g] = K(16 keys x 64d) @ Q^T : rows=keys, cols=queries ----
      v8f sg[2];
      #pragma unroll
      for (int g = 0; g < 2; ++g) {
        int krow = kk * 32 + g * 16 + l15;
        int aoff = hi ? 8 : 0;
        v8h a00 = *(const v8h*)&ks[krow * 64 + aoff];
        v8h a01 = *(const v8h*)&ks[krow * 64 + 16 + aoff];
        v16h ak0 = cat8(a00, a01);
        v8h a10 = *(const v8h*)&ks[krow * 64 + 32 + aoff];
        v8h a11 = *(const v8h*)&ks[krow * 64 + 48 + aoff];
        v16h ak1 = cat8(a10, a11);
        v8f s0 = {};
        s0 = __builtin_amdgcn_wmma_f32_16x16x32_f16(false, ak0, false, bq[0], (short)0, s0, false, false);
        s0 = __builtin_amdgcn_wmma_f32_16x16x32_f16(false, ak1, false, bq[1], (short)0, s0, false, false);
        sg[g] = s0;
      }
      // ---- boundary-only causal masking ----
      if (kaTile + 31 > qaMin) {
        #pragma unroll
        for (int g = 0; g < 2; ++g)
          #pragma unroll
          for (int i = 0; i < 8; ++i) {
            int ka = kaTile + g * 16 + i + (hi ? 8 : 0);
            if (ka > qa) sg[g][i] = -1e30f;
          }
      }
      // ---- online softmax (key = VGPR idx, query = lane) ----
      float rx = -1e30f;
      #pragma unroll
      for (int g = 0; g < 2; ++g)
        #pragma unroll
        for (int i = 0; i < 8; ++i) rx = fmaxf(rx, sg[g][i]);
      rx = fmaxf(rx, __shfl_xor(rx, 16, 32));
      float mnew = fmaxf(mrow, rx);
      float alpha = __expf(mrow - mnew);
      mrow = mnew;
      float rs = 0.f;
      #pragma unroll
      for (int g = 0; g < 2; ++g)
        #pragma unroll
        for (int i = 0; i < 8; ++i) {
          float p = __expf(sg[g][i] - mnew);
          sg[g][i] = p;
          rs += p;
        }
      rs += __shfl_xor(rs, 16, 32);
      lrow = lrow * alpha + rs;
      #pragma unroll
      for (int fd = 0; fd < 4; ++fd)
        #pragma unroll
        for (int i = 0; i < 8; ++i) oacc[fd][i] *= alpha;

      // ---- P^T -> per-wave LDS [q][key32] (wave-local: no block barrier) ----
      {
        _Float16* pw = ps + wv * 512;
        v8h pk0, pk1;
        #pragma unroll
        for (int i = 0; i < 8; ++i) { pk0[i] = (_Float16)sg[0][i]; pk1[i] = (_Float16)sg[1][i]; }
        int off = hi ? 8 : 0;
        *(v8h*)&pw[l15 * 32 + off]      = pk0;
        *(v8h*)&pw[l15 * 32 + 16 + off] = pk1;
      }
      wave_lds_sync();
      // ---- O^T += V^T(16d x 32keys) @ P^T(32keys x 16q) ----
      v16h bp;
      {
        const _Float16* pw = ps + wv * 512;
        int off = hi ? 8 : 0;
        v8h p0 = *(const v8h*)&pw[l15 * 32 + off];
        v8h p1 = *(const v8h*)&pw[l15 * 32 + off + 16];
        bp = cat8(p0, p1);
      }
      #pragma unroll
      for (int fd = 0; fd < 4; ++fd) {
        int drow = fd * 16 + l15;
        int aoff = kk * 32 + (hi ? 8 : 0);
        v8h q0 = *(const v8h*)&vs[drow * 64 + aoff];
        v8h q1 = *(const v8h*)&vs[drow * 64 + 16 + aoff];
        v16h av = cat8(q0, q1);
        oacc[fd] = __builtin_amdgcn_wmma_f32_16x16x32_f16(false, av, false, bp, (short)0, oacc[fd], false, false);
      }
      wave_lds_sync();                       // protect ps before next chunk's stores
    }
  }

  // ---- finalize: O^T / l ; packed b128 stores (lane = query/token) ----
  float rinv = 1.0f / lrow;
  int tok = tok0 + wv * 16 + l15;
  size_t obase = (rowbase + tok) * (size_t)(H_ * DH_) + h * DH_;
  #pragma unroll
  for (int fd = 0; fd < 4; ++fd) {
    v8h o;
    #pragma unroll
    for (int i = 0; i < 8; ++i) o[i] = (_Float16)(oacc[fd][i] * rinv);
    *(v8h*)&outh[obase + fd * 16 + (hi ? 8 : 0)] = o;
  }
}

// ------------------------------------------------ host
extern "C" void kernel_launch(void* const* d_in, const int* in_sizes, int n_in,
                              void* d_out, int out_size, void* d_ws, size_t ws_size,
                              hipStream_t stream) {
  (void)in_sizes; (void)n_in; (void)out_size; (void)ws_size;
  const float* x    = (const float*)d_in[0];
  const float* ln1g = (const float*)d_in[1];
  const float* ln1b = (const float*)d_in[2];
  const float* Wqkv = (const float*)d_in[3];
  const float* Wout = (const float*)d_in[4];
  const float* ln2g = (const float*)d_in[5];
  const float* ln2b = (const float*)d_in[6];
  const float* W1   = (const float*)d_in[7];
  const float* b1   = (const float*)d_in[8];
  const float* W2   = (const float*)d_in[9];
  const float* b2   = (const float*)d_in[10];

  char* ws = (char*)d_ws;
  size_t off = 0;
  auto alloc = [&](size_t bytes) {
    void* p = ws + off;
    off = (off + bytes + 255) & ~(size_t)255;
    return p;
  };
  float*    xcur   = (float*)   alloc((size_t)M_ * D_ * 4);
  _Float16* h16    = (_Float16*)alloc((size_t)M_ * D_ * 2);
  _Float16* qkv16  = (_Float16*)alloc((size_t)M_ * 1536 * 2);        // rotated q,k + v
  _Float16* attno  = (_Float16*)alloc((size_t)M_ * 512 * 2);
  _Float16* ff116  = (_Float16*)alloc((size_t)M_ * DFF_ * 2);
  _Float16* wqkv16 = (_Float16*)alloc((size_t)L_ * D_ * 1536 * 2);   // [1536][512] per layer
  _Float16* wout16 = (_Float16*)alloc((size_t)L_ * 512 * 512 * 2);   // [512][512]
  _Float16* w116   = (_Float16*)alloc((size_t)L_ * D_ * DFF_ * 2);   // [2048][512]
  _Float16* w216   = (_Float16*)alloc((size_t)L_ * DFF_ * D_ * 2);   // [512][2048]

  (void)hipMemcpyAsync(xcur, x, (size_t)M_ * D_ * 4, hipMemcpyDeviceToDevice, stream);

  for (int l = 0; l < L_; ++l) {
    k_conv_tr<<<384, 256, 0, stream>>>(Wqkv + (size_t)l * D_ * 1536,
                                       wqkv16 + (size_t)l * D_ * 1536, D_, 1536);
    k_conv_tr<<<128, 256, 0, stream>>>(Wout + (size_t)l * 512 * 512,
                                       wout16 + (size_t)l * 512 * 512, 512, 512);
    k_conv_tr<<<512, 256, 0, stream>>>(W1 + (size_t)l * D_ * DFF_,
                                       w116 + (size_t)l * D_ * DFF_, D_, DFF_);
    k_conv_tr<<<512, 256, 0, stream>>>(W2 + (size_t)l * DFF_ * D_,
                                       w216 + (size_t)l * DFF_ * D_, DFF_, D_);
  }

  for (int l = 0; l < L_; ++l) {
    k_layernorm_f16<<<M_, 256, 0, stream>>>(xcur, ln1g + l * D_, ln1b + l * D_, h16);
    {
      dim3 g(1536 / 128, M_ / 128);   // EPI 3: rotary + scale fused, f16 out
      k_gemm_wmma<3><<<g, 256, 0, stream>>>(h16, wqkv16 + (size_t)l * D_ * 1536,
                                            nullptr, nullptr, qkv16, D_, 1536);
    }
    k_local_attn<<<512, 256, 0, stream>>>(qkv16, attno);
    {
      dim3 g(512 / 128, M_ / 128);
      k_gemm_wmma<1><<<g, 256, 0, stream>>>(attno, wout16 + (size_t)l * 512 * 512,
                                            nullptr, xcur, xcur, 512, 512);
    }
    k_layernorm_f16<<<M_, 256, 0, stream>>>(xcur, ln2g + l * D_, ln2b + l * D_, h16);
    {
      dim3 g(DFF_ / 128, M_ / 128);
      k_gemm_wmma<2><<<g, 256, 0, stream>>>(h16, w116 + (size_t)l * D_ * DFF_,
                                            b1 + l * DFF_, nullptr, ff116, D_, DFF_);
    }
    {
      dim3 g(512 / 128, M_ / 128);
      k_gemm_wmma<1><<<g, 256, 0, stream>>>(ff116, w216 + (size_t)l * DFF_ * D_,
                                            b2 + l * D_, xcur, xcur, DFF_, 512);
    }
  }
  (void)hipMemcpyAsync(d_out, xcur, (size_t)M_ * D_ * 4, hipMemcpyDeviceToDevice, stream);
}
